// LGMRec_29712583754279
// MI455X (gfx1250) — matrix operand order repaired and checked by reading, not compiled, not run on hardware
//
#include <hip/hip_runtime.h>

// ---------------- constants (match reference) ----------------
#define HIDW   128
#define HYPW   256
#define TAUINV 2.0f      // 1 / TAU, TAU = 0.5
#define ALPHAW 0.1f
#define EPSW   1e-10f

typedef __attribute__((ext_vector_type(16))) __bf16 v16bf;
typedef __attribute__((ext_vector_type(8)))  float  v8f;

static __device__ __forceinline__ __bf16 to_bf(float f) { return (__bf16)f; }

// ===========================================================================
// Fragment packing.
// A-frag tile (16x32, bf16): lane<16 -> m=lane,    K {0..7, 16..23}
//                            lane>=16 -> m=lane-16, K {8..15, 24..31}
// B-frag tile (32x16, bf16): n = lane&15; lanes 0-15 hold K {0..7, 16..23},
//                            lanes 16-31 hold K {8..15, 24..31}
// Tile storage: tile index t, lane l -> 16 bf16 at (v16bf*)buf[t*32 + l]
// A tiles ordered t = mt*KT + kt ; B tiles ordered t = nt*KT + kt
// ===========================================================================

// A row-major [M x K], K % 32 == 0 (all row-major A cases). Vector loads.
__global__ void pack_a_rm_kernel(const float* __restrict__ A, int K, int KT,
                                 v16bf* __restrict__ out, long total)
{
    long t = (long)blockIdx.x * blockDim.x + threadIdx.x;
    if (t >= total) return;                 // total = MT*KT*32
    const int lane = (int)(t & 31);
    const long tile = t >> 5;
    const int mt = (int)(tile / KT), kt = (int)(tile % KT);
    const int m  = mt * 16 + (lane & 15);
    const int kb = kt * 32 + (lane >> 4) * 8;
    const float* p = A + (long)m * K + kb;
    v16bf a;
    #pragma unroll
    for (int i = 0; i < 8; ++i) a[i]     = to_bf(p[i]);
    #pragma unroll
    for (int i = 0; i < 8; ++i) a[8 + i] = to_bf(p[16 + i]);
    out[tile * 32 + lane] = a;
}

// A transposed: logical A[m][k] = S[k*lds + m]  (for Hm^T). Branch-free k guard.
__global__ void pack_a_tr_kernel(const float* __restrict__ S, int lds, int K, int KT,
                                 v16bf* __restrict__ out, long total)
{
    long t = (long)blockIdx.x * blockDim.x + threadIdx.x;
    if (t >= total) return;
    const int lane = (int)(t & 31);
    const long tile = t >> 5;
    const int mt = (int)(tile / KT), kt = (int)(tile % KT);
    const int m  = mt * 16 + (lane & 15);
    const int ks = kt * 32 + (lane >> 4) * 8;
    v16bf a;
    #pragma unroll
    for (int i = 0; i < 8; ++i) {
        const int k1 = ks + i, k2 = ks + 16 + i;
        const int c1 = k1 < K ? k1 : K - 1, c2 = k2 < K ? k2 : K - 1;
        const float f1 = S[(long)c1 * lds + m];
        const float f2 = S[(long)c2 * lds + m];
        a[i]     = (k1 < K) ? to_bf(f1) : (__bf16)0.0f;
        a[8 + i] = (k2 < K) ? to_bf(f2) : (__bf16)0.0f;
    }
    out[tile * 32 + lane] = a;
}

// B row-major [K x Ncols], ldn = Ncols. Branch-free k guard (K=50000 case).
__global__ void pack_b_kernel(const float* __restrict__ B, int ldn, int K, int KT,
                              v16bf* __restrict__ out, long total)
{
    long t = (long)blockIdx.x * blockDim.x + threadIdx.x;
    if (t >= total) return;                 // total = NT*KT*32
    const int lane = (int)(t & 31);
    const long tile = t >> 5;
    const int nt = (int)(tile / KT), kt = (int)(tile % KT);
    const int n  = nt * 16 + (lane & 15);
    const int ks = kt * 32 + (lane >> 4) * 8;
    v16bf b;
    #pragma unroll
    for (int i = 0; i < 16; ++i) {
        const int k = ks + (i < 8 ? i : i + 8);
        const int kc = k < K ? k : K - 1;
        const float f = B[(long)kc * ldn + n];
        b[i] = (k < K) ? to_bf(f) : (__bf16)0.0f;
    }
    out[tile * 32 + lane] = b;
}

// ===========================================================================
// GEMM on packed fragments: tight loop = 4x b128 load + 1x wmma per k-tile.
// block = 256 threads = 8 waves; wave w -> ntile blockIdx.y*8+w, mtile blockIdx.x
// ===========================================================================
__global__ void wmma_gemm_packed_kernel(const v16bf* __restrict__ pA,
                                        const v16bf* __restrict__ pB,
                                        int KT, const float* __restrict__ bias,
                                        float* __restrict__ C, long ldc,
                                        int Ncols, int relu)
{
    const int lane = threadIdx.x & 31;
    const int wave = threadIdx.x >> 5;
    const int nt   = blockIdx.y * 8 + wave;
    if (nt * 16 >= Ncols) return;           // wave-uniform
    const int mt = blockIdx.x;
    const v16bf* Ap = pA + (long)mt * KT * 32 + lane;
    const v16bf* Bp = pB + (long)nt * KT * 32 + lane;

    v8f c = {};
    #pragma unroll 4
    for (int kt = 0; kt < KT; ++kt)
        c = __builtin_amdgcn_wmma_f32_16x16x32_bf16(false, Ap[(long)kt * 32],
                                                    false, Bp[(long)kt * 32],
                                                    (short)0, c, false, false);

    const int half = lane >> 4, lm = lane & 15;
    const int n = nt * 16 + lm;
    const float bv = bias ? bias[n] : 0.0f;
    #pragma unroll
    for (int i = 0; i < 8; ++i) {
        const int row = mt * 16 + half * 8 + i;     // C layout: M = vgpr + 8*half
        float v = c[i] + bv;
        if (relu) v = fmaxf(v, 0.0f);
        C[(long)row * ldc + n] = v;
    }
}

// split-K over k-tiles with f32 atomic reduction (for lat = Hm^T @ x_emb)
__global__ void wmma_gemm_packed_splitk_kernel(const v16bf* __restrict__ pA,
                                               const v16bf* __restrict__ pB,
                                               int KT, int ktchunk,
                                               float* __restrict__ C, long ldc,
                                               int Ncols)
{
    const int lane = threadIdx.x & 31;
    const int wave = threadIdx.x >> 5;
    const int nt   = wave;
    if (nt * 16 >= Ncols) return;
    const int kt0 = blockIdx.z * ktchunk;
    if (kt0 >= KT) return;
    const int kt1 = min(KT, kt0 + ktchunk);
    const int mt = blockIdx.x;
    const v16bf* Ap = pA + (long)mt * KT * 32 + lane;
    const v16bf* Bp = pB + (long)nt * KT * 32 + lane;

    v8f c = {};
    #pragma unroll 4
    for (int kt = kt0; kt < kt1; ++kt)
        c = __builtin_amdgcn_wmma_f32_16x16x32_bf16(false, Ap[(long)kt * 32],
                                                    false, Bp[(long)kt * 32],
                                                    (short)0, c, false, false);

    const int half = lane >> 4, lm = lane & 15;
    const int n = nt * 16 + lm;
    #pragma unroll
    for (int i = 0; i < 8; ++i) {
        const int row = mt * 16 + half * 8 + i;
        atomicAdd(&C[(long)row * ldc + n], c[i]);
    }
}

// ---------------- elementwise / graph kernels -----------------------------
__global__ void fill_kernel(float* __restrict__ p, long n, float v)
{
    long i = (long)blockIdx.x * blockDim.x + threadIdx.x;
    if (i < n) p[i] = v;
}

__global__ void deg_kernel(const int* __restrict__ dst, float* __restrict__ deg, int E)
{
    int e = blockIdx.x * blockDim.x + threadIdx.x;
    if (e < E) atomicAdd(&deg[dst[e]], 1.0f);
}

__global__ void dinv_kernel(float* __restrict__ deg, int n)
{
    int i = blockIdx.x * blockDim.x + threadIdx.x;
    if (i < n) { float d = deg[i]; deg[i] = (d > 0.0f) ? (1.0f / sqrtf(d)) : 0.0f; }
}

__global__ void ew_kernel(const int* __restrict__ src, const int* __restrict__ dst,
                          const float* __restrict__ dinv, float* __restrict__ ew, int E)
{
    int e = blockIdx.x * blockDim.x + threadIdx.x;
    if (e < E) ew[e] = dinv[src[e]] * dinv[dst[e]];
}

// one float4 per thread: nxt[dst] += ew * cur[src]
__global__ void scatter_kernel(const int* __restrict__ src, const int* __restrict__ dst,
                               const float* __restrict__ ew,
                               const float* __restrict__ cur, float* __restrict__ nxt, int E)
{
    long t = (long)blockIdx.x * blockDim.x + threadIdx.x;
    if (t >= (long)E * (HIDW / 4)) return;
    const int e = (int)(t >> 5);                  // HIDW/4 == 32 chunks per edge
    const int f = ((int)t & 31) * 4;
    const float w = ew[e];
    const float4 v = *(const float4*)(cur + (long)src[e] * HIDW + f);
    float* o = nxt + (long)dst[e] * HIDW + f;
    atomicAdd(o + 0, v.x * w);
    atomicAdd(o + 1, v.y * w);
    atomicAdd(o + 2, v.z * w);
    atomicAdd(o + 3, v.w * w);
}

__global__ void add_kernel(float* __restrict__ acc, const float* __restrict__ x, long n)
{
    long i = (long)blockIdx.x * blockDim.x + threadIdx.x;
    if (i < n) acc[i] += x[i];
}

// gumbel-softmax over HYP=256, one wave32 per row, in-place safe
__global__ void gumbel_softmax_kernel(const float* logits, const float* u,
                                      float* out, int Nn)
{
    const int wave = threadIdx.x >> 5, lane = threadIdx.x & 31;
    const int row = blockIdx.x * 8 + wave;
    if (row >= Nn) return;
    const float* lr = logits + (long)row * HYPW;
    const float* ur = u      + (long)row * HYPW;
    float z[8];
    float mx = -1e30f;
    #pragma unroll
    for (int j = 0; j < 8; ++j) {
        const int idx = lane + j * 32;
        const float g = -logf(-logf(ur[idx] + EPSW) + EPSW);
        z[j] = (lr[idx] + g) * TAUINV;
        mx = fmaxf(mx, z[j]);
    }
    #pragma unroll
    for (int s = 16; s > 0; s >>= 1) mx = fmaxf(mx, __shfl_xor(mx, s, 32));
    float sum = 0.0f;
    #pragma unroll
    for (int j = 0; j < 8; ++j) { z[j] = expf(z[j] - mx); sum += z[j]; }
    #pragma unroll
    for (int s = 16; s > 0; s >>= 1) sum += __shfl_xor(sum, s, 32);
    const float inv = 1.0f / sum;
    #pragma unroll
    for (int j = 0; j < 8; ++j) out[(long)row * HYPW + lane + j * 32] = z[j] * inv;
}

// final = acc/4 + ALPHA * hyper / max(||hyper||, 1e-12), one wave per row
__global__ void final_kernel(float* __restrict__ accf, const float* __restrict__ hyper, int Nn)
{
    const int wave = threadIdx.x >> 5, lane = threadIdx.x & 31;
    const int row = blockIdx.x * 8 + wave;
    if (row >= Nn) return;
    const float* h = hyper + (long)row * HIDW;
    float hv[4];
    float s = 0.0f;
    #pragma unroll
    for (int j = 0; j < 4; ++j) { hv[j] = h[lane + j * 32]; s += hv[j] * hv[j]; }
    #pragma unroll
    for (int t = 16; t > 0; t >>= 1) s += __shfl_xor(s, t, 32);
    const float inv = ALPHAW / fmaxf(sqrtf(s), 1e-12f);
    float* a = accf + (long)row * HIDW;
    #pragma unroll
    for (int j = 0; j < 4; ++j) {
        const int idx = lane + j * 32;
        a[idx] = a[idx] * 0.25f + hv[j] * inv;
    }
}

// ---------------- host launcher -------------------------------------------
extern "C" void kernel_launch(void* const* d_in, const int* in_sizes, int n_in,
                              void* d_out, int out_size, void* d_ws, size_t ws_size,
                              hipStream_t stream)
{
    (void)n_in; (void)out_size; (void)ws_size;
    const float* x      = (const float*)d_in[0];
    const int*   edge   = (const int*)  d_in[1];
    const float* gum    = (const float*)d_in[2];
    const float* w_feat = (const float*)d_in[3];
    const float* b_feat = (const float*)d_in[4];
    const float* w_hyp  = (const float*)d_in[5];
    const float* w_vis  = (const float*)d_in[6];
    const float* b_vis  = (const float*)d_in[7];
    const float* w_txt  = (const float*)d_in[8];
    const float* b_txt  = (const float*)d_in[9];

    const int E      = in_sizes[1] / 2;
    const int N      = in_sizes[2] / HYPW;     // 50000
    const int IN_DIM = in_sizes[0] / N;        // 384
    const int* src = edge;
    const int* dst = edge + E;

    const int MT  = N / 16;                    // 3125 (N % 16 == 0)
    const int KTn = (N + 31) / 32;             // 1563 k-tiles for K = N

    // workspace carve-up (256B aligned)
    char* wsb = (char*)d_ws;
    size_t off = 0;
    auto take = [&](size_t bytes) -> char* {
        char* p = wsb + off;
        off = (off + bytes + 255) & ~(size_t)255;
        return p;
    };
    float* dinv  = (float*)take((size_t)N * 4);            // deg -> dinv in place
    float* ew    = (float*)take((size_t)E * 4);
    float* x_emb = (float*)take((size_t)N * HIDW * 4);
    float* cur   = (float*)take((size_t)N * HIDW * 4);
    float* nxt   = (float*)take((size_t)N * HIDW * 4);
    float* Hm    = (float*)take((size_t)N * HYPW * 4);     // logits -> Hm in place
    float* lat   = (float*)take((size_t)HYPW * HIDW * 4);
    float* hyper = (float*)take((size_t)N * HIDW * 4);
    // fragment buffers, reused across GEMMs (sizes = max over all uses)
    v16bf* pA = (v16bf*)take((size_t)MT * (IN_DIM / 32) * 1024);   // 38.4 MB
    v16bf* pB = (v16bf*)take((size_t)(HIDW / 16) * KTn * 1024);    // 12.8 MB

    float* acc = (float*)d_out;                // final slice doubles as acc
    float* vis = acc + (size_t)N * HIDW;
    float* txt = vis + (size_t)N * HIDW;

    const long nh = (long)N * HIDW;
    const dim3 blk(256);
    auto cdiv = [](long a, long b) -> long { return (a + b - 1) / b; };

    // 1) edge weights: deg -> dinv -> ew
    fill_kernel<<<cdiv(N, 256), blk, 0, stream>>>(dinv, N, 0.0f);
    deg_kernel <<<cdiv(E, 256), blk, 0, stream>>>(dst, dinv, E);
    dinv_kernel<<<cdiv(N, 256), blk, 0, stream>>>(dinv, N);
    ew_kernel  <<<cdiv(E, 256), blk, 0, stream>>>(src, dst, dinv, ew, E);

    // 2) x_emb = x @ w_feat + b_feat        (M=N, N=128, K=384)
    {
        const int KT = IN_DIM / 32;            // 12
        long ta = (long)MT * KT * 32, tb = (long)(HIDW / 16) * KT * 32;
        pack_a_rm_kernel<<<cdiv(ta, 256), blk, 0, stream>>>(x, IN_DIM, KT, pA, ta);
        pack_b_kernel   <<<cdiv(tb, 256), blk, 0, stream>>>(w_feat, HIDW, IN_DIM, KT, pB, tb);
        wmma_gemm_packed_kernel<<<dim3(MT, 1), blk, 0, stream>>>(
            pA, pB, KT, b_feat, x_emb, HIDW, HIDW, 0);
    }

    // 3) LightGCN: acc = x_emb + sum of 3 propagated layers
    hipMemcpyAsync(cur, x_emb, nh * 4, hipMemcpyDeviceToDevice, stream);
    hipMemcpyAsync(acc, x_emb, nh * 4, hipMemcpyDeviceToDevice, stream);
    const long st = (long)E * (HIDW / 4);
    for (int l = 0; l < 3; ++l) {
        fill_kernel   <<<cdiv(nh, 256), blk, 0, stream>>>(nxt, nh, 0.0f);
        scatter_kernel<<<cdiv(st, 256), blk, 0, stream>>>(src, dst, ew, cur, nxt, E);
        add_kernel    <<<cdiv(nh, 256), blk, 0, stream>>>(acc, nxt, nh);
        float* t = cur; cur = nxt; nxt = t;
    }

    // 4) logits = x_emb @ w_hyper           (M=N, N=256, K=128), then gumbel-softmax
    {
        const int KT = HIDW / 32;              // 4
        long ta = (long)MT * KT * 32, tb = (long)(HYPW / 16) * KT * 32;
        pack_a_rm_kernel<<<cdiv(ta, 256), blk, 0, stream>>>(x_emb, HIDW, KT, pA, ta);
        pack_b_kernel   <<<cdiv(tb, 256), blk, 0, stream>>>(w_hyp, HYPW, HIDW, KT, pB, tb);
        wmma_gemm_packed_kernel<<<dim3(MT, 2), blk, 0, stream>>>(
            pA, pB, KT, nullptr, Hm, HYPW, HYPW, 0);
    }
    gumbel_softmax_kernel<<<cdiv(N, 8), blk, 0, stream>>>(Hm, gum, Hm, N);

    // 5) lat = Hm^T @ x_emb                 (M=256, N=128, K=N) split-K + atomics
    {
        const int KT = KTn;                    // 1563 (tail zero-padded in packs)
        long ta = (long)(HYPW / 16) * KT * 32, tb = (long)(HIDW / 16) * KT * 32;
        pack_a_tr_kernel<<<cdiv(ta, 256), blk, 0, stream>>>(Hm, HYPW, N, KT, pA, ta);
        pack_b_kernel   <<<cdiv(tb, 256), blk, 0, stream>>>(x_emb, HIDW, N, KT, pB, tb);
        fill_kernel<<<cdiv(HYPW * HIDW, 256), blk, 0, stream>>>(lat, HYPW * HIDW, 0.0f);
        const int ktchunk = 64;
        const int nchunks = (KT + ktchunk - 1) / ktchunk;   // 25
        wmma_gemm_packed_splitk_kernel<<<dim3(HYPW / 16, 1, nchunks), blk, 0, stream>>>(
            pA, pB, KT, ktchunk, lat, HIDW, HIDW);
    }

    // 6) hyper = Hm @ lat                   (M=N, N=128, K=256)
    {
        const int KT = HYPW / 32;              // 8
        long ta = (long)MT * KT * 32, tb = (long)(HIDW / 16) * KT * 32;
        pack_a_rm_kernel<<<cdiv(ta, 256), blk, 0, stream>>>(Hm, HYPW, KT, pA, ta);
        pack_b_kernel   <<<cdiv(tb, 256), blk, 0, stream>>>(lat, HIDW, HYPW, KT, pB, tb);
        wmma_gemm_packed_kernel<<<dim3(MT, 1), blk, 0, stream>>>(
            pA, pB, KT, nullptr, hyper, HIDW, HIDW, 0);
    }

    // 7) final = acc/4 + alpha * normalize(hyper)   (in d_out slice 0)
    final_kernel<<<cdiv(N, 8), blk, 0, stream>>>(acc, hyper, N);

    // 8) x_vision / x_text = relu(final @ W + b)    (M=N, N=128, K=128)
    {
        const int KT = HIDW / 32;              // 4
        long ta = (long)MT * KT * 32, tb = (long)(HIDW / 16) * KT * 32;
        pack_a_rm_kernel<<<cdiv(ta, 256), blk, 0, stream>>>(acc, HIDW, KT, pA, ta);
        pack_b_kernel   <<<cdiv(tb, 256), blk, 0, stream>>>(w_vis, HIDW, HIDW, KT, pB, tb);
        wmma_gemm_packed_kernel<<<dim3(MT, 1), blk, 0, stream>>>(
            pA, pB, KT, b_vis, vis, HIDW, HIDW, 1);
        pack_b_kernel   <<<cdiv(tb, 256), blk, 0, stream>>>(w_txt, HIDW, HIDW, KT, pB, tb);
        wmma_gemm_packed_kernel<<<dim3(MT, 1), blk, 0, stream>>>(
            pA, pB, KT, b_txt, txt, HIDW, HIDW, 1);
    }
}
